// BiBatchHardTripletLoss_5145370820780
// MI455X (gfx1250) — matrix operand order repaired and checked by reading, not compiled, not run on hardware
//
#include <hip/hip_runtime.h>
#include <hip/hip_bf16.h>
#include <stdint.h>

// ---------------------------------------------------------------------------
// BiBatchHardTripletLoss for MI455X (gfx1250, wave32, WMMA)
//   rgb = inputs[:4096], ir = inputs[4096:], K=1024
//   dist[i][j] = sqrt(clip(|rgb_i|^2 + |ir_j|^2 - 2*rgb_i.ir_j, 1e-12))
//   cond(i,j)  = (tgt[j] == tgt[4096+i])       (exactly the reference's mask)
//   rowmax/rowmin over j, colmax/colmin over i, then two means of relu(0.3-(an-ap)).
// GEMM runs as bf16x3 split (hi*hi + hi*lo + lo*hi) on v_wmma_f32_16x16x32_bf16.
// Global loads are shaped as  SGPR64 base (+ compile-time imm) + zext(32-bit
// VGPR byte offset)  so the backend selects GVS addressing; the k-loop carries
// only two 32-bit offset adds.
// ---------------------------------------------------------------------------

typedef __attribute__((ext_vector_type(16))) __bf16 v16bf;
typedef __attribute__((ext_vector_type(8)))  float  v8f;

union Frag { uint4 u[2]; v16bf v; };

#define NHALF 4096
#define KDIM  1024
#define NEG_INF_BITS 0xFF800000u
#define POS_INF_BITS 0x7F800000u

__device__ __forceinline__ uint32_t bf16_rne(float x) {
  uint32_t u = __float_as_uint(x);
  return (u + 0x7FFFu + ((u >> 16) & 1u)) >> 16;
}

// addr = SGPR(base) + cbytes(imm24) + zext(VGPR byte offset)   -> GVS form
__device__ __forceinline__ ushort4 ldg4b(const uint16_t* __restrict__ p,
                                         int cbytes, uint32_t boff) {
  return *(const ushort4*)((const char*)p + cbytes + (size_t)boff);
}

// -------- Kernel A: fp32 -> (bf16 hi, bf16 lo) split + per-row squared norms
__global__ void prep_kernel(const float* __restrict__ x,
                            uint16_t* __restrict__ rgb_hi, uint16_t* __restrict__ rgb_lo,
                            uint16_t* __restrict__ ir_hi,  uint16_t* __restrict__ ir_lo,
                            float* __restrict__ rgbsq, float* __restrict__ irsq) {
  __shared__ float red[256];
  const int r = blockIdx.x, t = threadIdx.x;
  const float* row = x + (size_t)r * KDIM;
  uint16_t* hi; uint16_t* lo;
  if (r < NHALF) { hi = rgb_hi + (size_t)r * KDIM;           lo = rgb_lo + (size_t)r * KDIM; }
  else           { hi = ir_hi  + (size_t)(r - NHALF) * KDIM; lo = ir_lo + (size_t)(r - NHALF) * KDIM; }
  float s = 0.f;
  for (int c = t; c < KDIM; c += 256) {
    float v = row[c];
    s += v * v;
    uint32_t h  = bf16_rne(v);
    float    hf = __uint_as_float(h << 16);
    uint32_t l  = bf16_rne(v - hf);
    hi[c] = (uint16_t)h;
    lo[c] = (uint16_t)l;
  }
  red[t] = s; __syncthreads();
  for (int off = 128; off > 0; off >>= 1) {
    if (t < off) red[t] += red[t + off];
    __syncthreads();
  }
  if (t == 0) { if (r < NHALF) rgbsq[r] = red[0]; else irsq[r - NHALF] = red[0]; }
}

// -------- Kernel B: init reduction buffers to +-inf bit patterns
__global__ void init_kernel(int* __restrict__ rowmax, int* __restrict__ rowmin,
                            int* __restrict__ colmax, int* __restrict__ colmin) {
  int i = blockIdx.x * blockDim.x + threadIdx.x;
  if (i < NHALF) {
    rowmax[i] = (int)NEG_INF_BITS; rowmin[i] = (int)POS_INF_BITS;
    colmax[i] = (int)NEG_INF_BITS; colmin[i] = (int)POS_INF_BITS;
  }
}

// -------- Kernel C: fused WMMA GEMM + distance + masked row/col max/min
__global__ void __launch_bounds__(256)
gemm_reduce_kernel(const uint16_t* __restrict__ rgb_hi, const uint16_t* __restrict__ rgb_lo,
                   const uint16_t* __restrict__ ir_hi,  const uint16_t* __restrict__ ir_lo,
                   const float* __restrict__ rgbsq, const float* __restrict__ irsq,
                   const int* __restrict__ tgt,
                   int* __restrict__ rowmax, int* __restrict__ rowmin,
                   int* __restrict__ colmax, int* __restrict__ colmin) {
  __shared__ uint16_t sAh[128 * 32];
  __shared__ uint16_t sAl[128 * 32];
  __shared__ uint16_t sBh[128 * 32];
  __shared__ uint16_t sBl[128 * 32];

  const int tid  = threadIdx.x;
  const int bx   = blockIdx.x, by = blockIdx.y;
  const int lrow = tid >> 3;          // 0..31
  const int kcol = (tid & 7) << 2;    // 0,4,...,28  (bf16 elements)
  const int lane = tid & 31;
  const int wid  = tid >> 5;          // 8 waves
  const int wm   = wid >> 2;          // 0..1  -> 64 rows each
  const int wn   = wid & 3;           // 0..3  -> 32 cols each

  // Per-thread BYTE offsets (32-bit) for GVS addressing. The per-p tile stride
  // (32 rows * KDIM * 2B = 65536 B) is a compile-time pointer-side constant
  // that folds into the instruction's imm24 offset.
  const uint32_t aByte = ((uint32_t)(by * 128 + lrow) * KDIM + (uint32_t)kcol) * 2u;
  const uint32_t bByte = ((uint32_t)(bx * 128 + lrow) * KDIM + (uint32_t)kcol) * 2u;
  const int      PSTR  = 32 * KDIM * 2;      // 65536 bytes
  const int      sOff  = lrow * 32 + kcol;   // LDS store offset (p adds 32*32)

  v8f acc[4][2];
  {
    v8f z = {0.f, 0.f, 0.f, 0.f, 0.f, 0.f, 0.f, 0.f};
#pragma unroll
    for (int mi = 0; mi < 4; ++mi)
#pragma unroll
      for (int ni = 0; ni < 2; ++ni) acc[mi][ni] = z;
  }

  ushort4 stAh[4], stAl[4], stBh[4], stBl[4];
#pragma unroll
  for (int p = 0; p < 4; ++p) {   // preload k-step 0
    stAh[p] = ldg4b(rgb_hi, p * PSTR, aByte);
    stAl[p] = ldg4b(rgb_lo, p * PSTR, aByte);
    stBh[p] = ldg4b(ir_hi,  p * PSTR, bByte);
    stBl[p] = ldg4b(ir_lo,  p * PSTR, bByte);
  }

  const int NK = KDIM / 32;
  uint32_t koff = 0;                // +64 bytes (32 bf16) per k-step
  for (int ks = 0; ks < NK; ++ks) {
    __syncthreads();
#pragma unroll
    for (int p = 0; p < 4; ++p) {
      int off = sOff + p * (32 * 32);
      *(ushort4*)(sAh + off) = stAh[p];
      *(ushort4*)(sAl + off) = stAl[p];
      *(ushort4*)(sBh + off) = stBh[p];
      *(ushort4*)(sBl + off) = stBl[p];
    }
    __syncthreads();
    koff += 64u;
    if (ks + 1 < NK) {              // prefetch next k-step while WMMAs run
      uint32_t ra = aByte + koff;   // one 32-bit add each
      uint32_t rb = bByte + koff;
#pragma unroll
      for (int p = 0; p < 4; ++p) {
        stAh[p] = ldg4b(rgb_hi, p * PSTR, ra);
        stAl[p] = ldg4b(rgb_lo, p * PSTR, ra);
        stBh[p] = ldg4b(ir_hi,  p * PSTR, rb);
        stBl[p] = ldg4b(ir_lo,  p * PSTR, rb);
      }
    }

    // A fragment (16x32 bf16): lanes 0-15 row=l, K {0..7,16..23}; lanes 16-31 row=l-16, K {8..15,24..31}
    const int r15 = lane & 15;
    const int ka  = (lane < 16) ? 0 : 8;
    Frag aH[4], aL[4];
#pragma unroll
    for (int mi = 0; mi < 4; ++mi) {
      const uint16_t* pa = sAh + (wm * 64 + mi * 16 + r15) * 32 + ka;
      const uint16_t* pl = sAl + (wm * 64 + mi * 16 + r15) * 32 + ka;
      aH[mi].u[0] = *(const uint4*)(pa);
      aH[mi].u[1] = *(const uint4*)(pa + 16);
      aL[mi].u[0] = *(const uint4*)(pl);
      aL[mi].u[1] = *(const uint4*)(pl + 16);
    }
    // B fragment (32x16 bf16): lane n holds column n; lanes 0-15 K {0..7,16..23}, lanes 16-31 K {8..15,24..31}
    const int kb0 = (lane < 16) ? 0 : 8;
#pragma unroll
    for (int ni = 0; ni < 2; ++ni) {
      const uint16_t* pb = sBh + (wn * 32 + ni * 16 + r15) * 32 + kb0;
      const uint16_t* pc = sBl + (wn * 32 + ni * 16 + r15) * 32 + kb0;
      Frag bH, bL;
      bH.u[0] = *(const uint4*)(pb);
      bH.u[1] = *(const uint4*)(pb + 16);
      bL.u[0] = *(const uint4*)(pc);
      bL.u[1] = *(const uint4*)(pc + 16);
#pragma unroll
      for (int mi = 0; mi < 4; ++mi) {
        acc[mi][ni] = __builtin_amdgcn_wmma_f32_16x16x32_bf16(
            false, aH[mi].v, false, bH.v, (short)0, acc[mi][ni], false, false);
        acc[mi][ni] = __builtin_amdgcn_wmma_f32_16x16x32_bf16(
            false, aH[mi].v, false, bL.v, (short)0, acc[mi][ni], false, false);
        acc[mi][ni] = __builtin_amdgcn_wmma_f32_16x16x32_bf16(
            false, aL[mi].v, false, bH.v, (short)0, acc[mi][ni], false, false);
      }
    }
  }

  // ---- Epilogue: distances, mask, in-tile reductions, monotonic atomics ----
  const int   half = lane >> 4;       // C layout: VGPR e holds rows e (lanes 0-15) / e+8 (16-31)
  const int   l15  = lane & 15;
  const int   rowBase = by * 128 + wm * 64;
  const int   colBase = bx * 128 + wn * 32;
  const float INF = __int_as_float((int)POS_INF_BITS);
  const int* __restrict__ tgt_ir = tgt + NHALF;

  float irsqv[2]; int clbl[2];
#pragma unroll
  for (int ni = 0; ni < 2; ++ni) {
    int j = colBase + ni * 16 + l15;
    irsqv[ni] = irsq[j];
    clbl[ni]  = tgt[j];                // reference mask uses tgt[col] ...
  }
  float cMax[2] = {-INF, -INF};
  float cMin[2] = { INF,  INF};

#pragma unroll
  for (int mi = 0; mi < 4; ++mi) {
#pragma unroll
    for (int e = 0; e < 8; ++e) {
      int   i  = rowBase + mi * 16 + half * 8 + e;
      float rq = rgbsq[i];
      int   rl = tgt_ir[i];            // ... against tgt[4096 + row]
      float vM = -INF, vm = INF;
#pragma unroll
      for (int ni = 0; ni < 2; ++ni) {
        float dot = acc[mi][ni][e];
        float sq  = rq + irsqv[ni] - 2.0f * dot;
        float d   = sqrtf(fmaxf(sq, 1e-12f));
        bool  cond = (clbl[ni] == rl);
        float dM = cond ? d : -INF;
        float dm = cond ? INF : d;
        vM = fmaxf(vM, dM);  vm = fminf(vm, dm);
        cMax[ni] = fmaxf(cMax[ni], dM);
        cMin[ni] = fminf(cMin[ni], dm);
      }
#pragma unroll
      for (int off = 1; off < 16; off <<= 1) {   // fold 16 lanes of this row
        vM = fmaxf(vM, __shfl_xor(vM, off, 32));
        vm = fminf(vm, __shfl_xor(vm, off, 32));
      }
      if (l15 == 0) {   // positive floats: int-bit order == float order (deterministic)
        atomicMax(rowmax + i, __float_as_int(vM));
        atomicMin(rowmin + i, __float_as_int(vm));
      }
    }
  }
#pragma unroll
  for (int ni = 0; ni < 2; ++ni) {
    float cM = fmaxf(cMax[ni], __shfl_xor(cMax[ni], 16, 32));
    float cm = fminf(cMin[ni], __shfl_xor(cMin[ni], 16, 32));
    if (half == 0) {
      int j = colBase + ni * 16 + l15;
      atomicMax(colmax + j, __float_as_int(cM));
      atomicMin(colmin + j, __float_as_int(cm));
    }
  }
}

// -------- Kernel D: deterministic final mean of the two hinge terms
__global__ void finalize_kernel(const int* __restrict__ rowmax, const int* __restrict__ rowmin,
                                const int* __restrict__ colmax, const int* __restrict__ colmin,
                                float* __restrict__ out) {
  __shared__ float red[256];
  const int t = threadIdx.x;
  float s = 0.f;
  for (int i = t; i < NHALF; i += 256) {
    float ap = __int_as_float(rowmax[i]);
    float an = __int_as_float(rowmin[i]);
    s += fmaxf(0.3f - (an - ap), 0.f);     // relu(-inf) == 0 matches jnp.where semantics
    float ap2 = __int_as_float(colmax[i]);
    float an2 = __int_as_float(colmin[i]);
    s += fmaxf(0.3f - (an2 - ap2), 0.f);
  }
  red[t] = s; __syncthreads();
  for (int off = 128; off > 0; off >>= 1) {
    if (t < off) red[t] += red[t + off];
    __syncthreads();
  }
  if (t == 0) out[0] = red[0] / (float)NHALF;   // meanR + meanC, both /4096
}

extern "C" void kernel_launch(void* const* d_in, const int* in_sizes, int n_in,
                              void* d_out, int out_size, void* d_ws, size_t ws_size,
                              hipStream_t stream) {
  (void)in_sizes; (void)n_in; (void)out_size; (void)ws_size;
  const float* inputs  = (const float*)d_in[0];
  const int*   targets = (const int*)d_in[1];

  char* ws = (char*)d_ws;                               // needs ~33.7 MB
  const size_t SZH = (size_t)NHALF * KDIM * sizeof(uint16_t);   // 8 MB each
  uint16_t* rgb_hi = (uint16_t*)(ws);
  uint16_t* rgb_lo = (uint16_t*)(ws + SZH);
  uint16_t* ir_hi  = (uint16_t*)(ws + 2 * SZH);
  uint16_t* ir_lo  = (uint16_t*)(ws + 3 * SZH);
  float* rgbsq = (float*)(ws + 4 * SZH);
  float* irsq  = rgbsq + NHALF;
  int* rowmax = (int*)(irsq + NHALF);
  int* rowmin = rowmax + NHALF;
  int* colmax = rowmin + NHALF;
  int* colmin = colmax + NHALF;

  prep_kernel<<<2 * NHALF, 256, 0, stream>>>(inputs, rgb_hi, rgb_lo, ir_hi, ir_lo, rgbsq, irsq);
  init_kernel<<<NHALF / 256, 256, 0, stream>>>(rowmax, rowmin, colmax, colmin);
  gemm_reduce_kernel<<<dim3(32, 32), 256, 0, stream>>>(rgb_hi, rgb_lo, ir_hi, ir_lo,
                                                       rgbsq, irsq, targets,
                                                       rowmax, rowmin, colmax, colmin);
  finalize_kernel<<<1, 256, 0, stream>>>(rowmax, rowmin, colmax, colmin, (float*)d_out);
}
